// LoFTREncoderLayer_29557964931175
// MI455X (gfx1250) — matrix operand order repaired and checked by reading, not compiled, not run on hardware
//
#include <hip/hip_runtime.h>
#include <hip/hip_bf16.h>
#include <math.h>

#define BB   16
#define NN   4800
#define DD   256
#define HH   8
#define DHH  32
#define RR   (BB*NN)          // 76800 rows
#define D2   512

typedef __bf16 bf16t;
typedef __attribute__((ext_vector_type(16))) __bf16 v16bf;
typedef __attribute__((ext_vector_type(8)))  __bf16 v8bf;
typedef __attribute__((ext_vector_type(8)))  float  v8f;

union FragA { v16bf v; v8bf h[2]; };

// A-fragment (16x32 bf16): lane m=L&15, half=L>>4.
// elements 0..7  = K = k0 + 16*half + 0..7  (VGPR 0..3)
// elements 8..15 = K = k0 + 16*half + 8..15 (VGPR 4..7)
__device__ __forceinline__ v16bf load_a_frag(const bf16t* rowk, int hi) {
  FragA u;
  u.h[0] = *(const v8bf*)(rowk + 16 * hi);
  u.h[1] = *(const v8bf*)(rowk + 16 * hi + 8);
  return u.v;
}

__device__ __forceinline__ v8f wmma_bf16(v16bf a, v16bf b, v8f c) {
  return __builtin_amdgcn_wmma_f32_16x16x32_bf16(false, a, false, b, (short)0, c,
                                                 false, false);
}

__device__ __forceinline__ v8f vzero8() {
  v8f z = {0.f, 0.f, 0.f, 0.f, 0.f, 0.f, 0.f, 0.f};
  return z;
}

// ---------------- elementwise converts ----------------
__global__ void f2b_kernel(const float* __restrict__ s, bf16t* __restrict__ d, size_t n) {
  size_t i = (size_t)blockIdx.x * blockDim.x + threadIdx.x;
  if (i < n) d[i] = (bf16t)s[i];
}

// W (K x Nc, row-major f32) -> Wt (col-major bf16: element (k,n) at n*K+k)
__global__ void wconv_kernel(const float* __restrict__ W, bf16t* __restrict__ Wt,
                             int K, int Nc) {
  int i = blockIdx.x * 256 + threadIdx.x;
  if (i < K * Nc) {
    int n = i % Nc;
    int k = i / Nc;
    Wt[(size_t)n * K + k] = (bf16t)W[i];
  }
}

// ---------------- generic bf16 WMMA GEMM ----------------
// C[R x Nc] = act( A[R x K] @ W[K x Nc] + bias ),  A split at ksplit into A0/A1.
// Wt column-major bf16. block=256 (8 waves); each wave: 32 rows x 64 cols
// (2 row-tiles x 4 col-tiles = 8 WMMAs per k-step, register double-buffered).
// ACT: 0=none, 1=elu+1, 2=relu.
template <int ACT, bool HASB, bool HASF>
__global__ void gemm_wmma(const bf16t* __restrict__ A0, const bf16t* __restrict__ A1,
                          int ksplit, int lda0, int lda1,
                          const bf16t* __restrict__ Wt, const float* __restrict__ bias,
                          bf16t* __restrict__ outB, float* __restrict__ outF,
                          int K, int Nc) {
  int wave = threadIdx.x >> 5;
  int lane = threadIdx.x & 31;
  int hi   = lane >> 4;
  int n16  = lane & 15;
  int rowbase = (blockIdx.x * 8 + wave) * 32;
  int colbase = blockIdx.y * 64;

  v8f acc[2][4];
#pragma unroll
  for (int rt = 0; rt < 2; ++rt)
#pragma unroll
    for (int t = 0; t < 4; ++t) acc[rt][t] = vzero8();

  auto aptr = [&](int k0, int rt) -> const bf16t* {
    return (k0 < ksplit)
        ? A0 + (size_t)(rowbase + rt * 16 + n16) * lda0 + k0
        : A1 + (size_t)(rowbase + rt * 16 + n16) * lda1 + (k0 - ksplit);
  };

  // prologue: fragments for k0 = 0
  v16bf a[2], b[4];
  a[0] = load_a_frag(aptr(0, 0), hi);
  a[1] = load_a_frag(aptr(0, 1), hi);
#pragma unroll
  for (int t = 0; t < 4; ++t)
    b[t] = *(const v16bf*)(Wt + (size_t)(colbase + t * 16 + n16) * K + 16 * hi);

  for (int k0 = 0; k0 < K; k0 += 32) {
    int kn = k0 + 32;
    v16bf an[2], bn[4];
    if (kn < K) {
      // issue next k-step's loads before consuming current fragments
      an[0] = load_a_frag(aptr(kn, 0), hi);
      an[1] = load_a_frag(aptr(kn, 1), hi);
#pragma unroll
      for (int t = 0; t < 4; ++t)
        bn[t] = *(const v16bf*)(Wt + (size_t)(colbase + t * 16 + n16) * K + kn + 16 * hi);
      int knn = k0 + 64;
      if (knn < K) {
        __builtin_prefetch(aptr(knn, 0), 0, 0);
        __builtin_prefetch(Wt + (size_t)(colbase + n16) * K + knn, 0, 0);
      }
    }
#pragma unroll
    for (int rt = 0; rt < 2; ++rt)
#pragma unroll
      for (int t = 0; t < 4; ++t)
        acc[rt][t] = wmma_bf16(a[rt], b[t], acc[rt][t]);
    if (kn < K) {
      a[0] = an[0]; a[1] = an[1];
#pragma unroll
      for (int t = 0; t < 4; ++t) b[t] = bn[t];
    }
  }

#pragma unroll
  for (int t = 0; t < 4; ++t) {
    int col = colbase + t * 16 + n16;
    float bv = bias[col];
#pragma unroll
    for (int rt = 0; rt < 2; ++rt)
#pragma unroll
      for (int r = 0; r < 8; ++r) {
        int row = rowbase + rt * 16 + r + 8 * hi;
        float v = acc[rt][t][r] + bv;
        if (ACT == 1) v = (v > 0.f) ? (v + 1.f) : __expf(v);   // elu(v)+1
        if (ACT == 2) v = (v > 0.f) ? v : 0.f;                 // relu
        if (HASB) outB[(size_t)row * Nc + col] = (bf16t)v;
        if (HASF) outF[(size_t)row * Nc + col] = v;
      }
  }
}

// ---------------- per-head transpose (B,M,H*32) -> (B,H,32,M) ----------------
__global__ void transpose_head(const bf16t* __restrict__ src, bf16t* __restrict__ dst) {
  __shared__ bf16t tile[32][33];
  int m0 = blockIdx.x * 32;
  int h = blockIdx.y, b = blockIdx.z;
  int t = threadIdx.x;
  int dl = t & 31, ml = t >> 5;
#pragma unroll
  for (int j = 0; j < 4; ++j) {
    int m = ml + j * 8;
    tile[m][dl] = src[((size_t)(b * NN + m0 + m)) * DD + h * DHH + dl];
  }
  __syncthreads();
  int ml2 = t & 31, dl2 = t >> 5;
#pragma unroll
  for (int j = 0; j < 4; ++j) {
    int d = dl2 + j * 8;
    dst[((size_t)((b * HH + h) * DHH + d)) * NN + m0 + ml2] = tile[ml2][d];
  }
}

// ---------------- kv^T = v^T @ k  per (b,h); K = M = 4800 ----------------
// A = vT (32_e x M, row-major), B = k (M x 32_d) supplied col-major as kT.
// block=128 (4 waves), one wave per 16x16 tile. grid = B*H.
__global__ void kv_wmma(const bf16t* __restrict__ vT, const bf16t* __restrict__ kT,
                        bf16t* __restrict__ kvT) {
  int bh = blockIdx.x;
  int wave = threadIdx.x >> 5, lane = threadIdx.x & 31;
  int hi = lane >> 4, n16 = lane & 15;
  int tr = wave >> 1, tc = wave & 1;
  const bf16t* Arow = vT + (size_t)bh * DHH * NN + (size_t)(tr * 16 + n16) * NN;
  const bf16t* Bcol = kT + (size_t)bh * DHH * NN + (size_t)(tc * 16 + n16) * NN;
  v8f acc = vzero8();
  v16bf a = load_a_frag(Arow, hi);
  v16bf b = *(const v16bf*)(Bcol + 16 * hi);
  for (int k0 = 0; k0 < NN; k0 += 32) {
    int kn = k0 + 32;
    v16bf an, bn;
    if (kn < NN) {
      an = load_a_frag(Arow + kn, hi);
      bn = *(const v16bf*)(Bcol + kn + 16 * hi);
    }
    acc = wmma_bf16(a, b, acc);
    if (kn < NN) { a = an; b = bn; }
  }
#pragma unroll
  for (int r = 0; r < 8; ++r) {
    int e = tr * 16 + r + 8 * hi;
    int d = tc * 16 + n16;
    kvT[(size_t)bh * (DHH * DHH) + e * DHH + d] = (bf16t)acc[r];
  }
}

// ---------------- z[b,h,d] = sum_m k_feat ---- one wave per row of kT ----------
__global__ void z_kernel(const bf16t* __restrict__ kT, float* __restrict__ z) {
  int row = blockIdx.x * 8 + (threadIdx.x >> 5);   // (b*H+h)*32 + d
  int lane = threadIdx.x & 31;
  const bf16t* p = kT + (size_t)row * NN;
  float s = 0.f;
  for (int m = lane; m < NN; m += 32) s += (float)p[m];
  for (int off = 16; off; off >>= 1) s += __shfl_xor(s, off, 32);
  if (lane == 0) z[row] = s;
}

// ---------------- normalizer[b,n,h] = q_row . z ----------------
__global__ void norm_kernel(const bf16t* __restrict__ q, const float* __restrict__ z,
                            float* __restrict__ nrm) {
  size_t i = (size_t)blockIdx.x * 256 + threadIdx.x;   // over B*N*H
  int h = (int)(i & 7);
  size_t bn = i >> 3;
  int b = (int)(bn / NN);
  const bf16t* qp = q + bn * DD + h * DHH;
  const float* zp = z + ((size_t)b * HH + h) * DHH;
  float s = 0.f;
#pragma unroll
  for (int d = 0; d < DHH; ++d) s += (float)qp[d] * zp[d];
  nrm[i] = s;
}

// ---------------- msg = (q @ kv) / (norm + 1e-6)  per (b,h), K = 32 ----------
// block=128 (4 waves); each wave: 16 rows x 32 cols. grid = (N/64, H, B).
__global__ void msg_wmma(const bf16t* __restrict__ q, const bf16t* __restrict__ kvT,
                         const float* __restrict__ nrm, bf16t* __restrict__ msg) {
  int b = blockIdx.z, h = blockIdx.y;
  int wave = threadIdx.x >> 5, lane = threadIdx.x & 31;
  int hi = lane >> 4, n16 = lane & 15;
  int rowbase = (blockIdx.x * 4 + wave) * 16;
  const bf16t* Abase = q + ((size_t)b * NN) * DD + h * DHH;
  const bf16t* Bbase = kvT + ((size_t)(b * HH + h)) * (DHH * DHH);  // [e*32+d]
  v8f acc[2];
  acc[0] = vzero8(); acc[1] = vzero8();
  v16bf a = load_a_frag(Abase + (size_t)(rowbase + n16) * DD, hi);
#pragma unroll
  for (int t = 0; t < 2; ++t) {
    v16bf bfr = *(const v16bf*)(Bbase + (t * 16 + n16) * DHH + 16 * hi);
    acc[t] = wmma_bf16(a, bfr, acc[t]);
  }
#pragma unroll
  for (int t = 0; t < 2; ++t)
#pragma unroll
    for (int r = 0; r < 8; ++r) {
      int row = rowbase + r + 8 * hi;
      float nm = nrm[((size_t)b * NN + row) * HH + h] + 1e-6f;
      msg[((size_t)(b * NN + row)) * DD + h * DHH + t * 16 + n16] =
          (bf16t)(acc[t][r] / nm);
    }
}

// ---------------- LayerNorm over 256 cols; one wave per row ----------------
__global__ void ln_kernel(const float* __restrict__ x, const float* __restrict__ res,
                          const float* __restrict__ g, const float* __restrict__ bta,
                          bf16t* __restrict__ outB, float* __restrict__ outF) {
  int row = blockIdx.x * 8 + (threadIdx.x >> 5);
  int lane = threadIdx.x & 31;
  const float* xp = x + (size_t)row * DD;
  float v[8], s = 0.f, ss = 0.f;
#pragma unroll
  for (int j = 0; j < 8; ++j) {
    float t = xp[lane + 32 * j];
    if (res) t += res[(size_t)row * DD + lane + 32 * j];
    v[j] = t; s += t; ss += t * t;
  }
  for (int off = 16; off; off >>= 1) {
    s += __shfl_xor(s, off, 32);
    ss += __shfl_xor(ss, off, 32);
  }
  float mu = s * (1.f / DD);
  float var = ss * (1.f / DD) - mu * mu;
  float rs = rsqrtf(var + 1e-6f);
#pragma unroll
  for (int j = 0; j < 8; ++j) {
    int c = lane + 32 * j;
    float o = (v[j] - mu) * rs * g[c] + bta[c];
    if (outB) outB[(size_t)row * DD + c] = (bf16t)o;
    if (outF) outF[(size_t)row * DD + c] = o;
  }
}

extern "C" void kernel_launch(void* const* d_in, const int* in_sizes, int n_in,
                              void* d_out, int out_size, void* d_ws, size_t ws_size,
                              hipStream_t stream) {
  const float* x   = (const float*)d_in[0];
  const float* src = (const float*)d_in[1];
  const float* Wq  = (const float*)d_in[2];
  const float* bq  = (const float*)d_in[3];
  const float* Wk  = (const float*)d_in[4];
  const float* bk  = (const float*)d_in[5];
  const float* Wv  = (const float*)d_in[6];
  const float* bv  = (const float*)d_in[7];
  const float* Wm  = (const float*)d_in[8];
  const float* bm  = (const float*)d_in[9];
  const float* g1  = (const float*)d_in[10];
  const float* b1  = (const float*)d_in[11];
  const float* W0  = (const float*)d_in[12];
  const float* b0  = (const float*)d_in[13];
  const float* W2  = (const float*)d_in[14];
  const float* b2  = (const float*)d_in[15];
  const float* g2  = (const float*)d_in[16];
  const float* b2n = (const float*)d_in[17];
  float* out = (float*)d_out;

  char* w = (char*)d_ws;
  const size_t S1 = (size_t)RR * DD * sizeof(bf16t);   // 39.3 MB
  bf16t* x_bf   = (bf16t*)(w + 0 * S1);
  bf16t* src_bf = (bf16t*)(w + 1 * S1);
  bf16t* q_bf   = (bf16t*)(w + 2 * S1);
  bf16t* k_bf   = (bf16t*)(w + 3 * S1);
  bf16t* v_bf   = (bf16t*)(w + 4 * S1);
  bf16t* kT     = (bf16t*)(w + 5 * S1);
  bf16t* vT     = (bf16t*)(w + 6 * S1);
  char* wp = w + 7 * S1;
  bf16t* WqT = (bf16t*)wp; wp += (size_t)DD * DD * 2;
  bf16t* WkT = (bf16t*)wp; wp += (size_t)DD * DD * 2;
  bf16t* WvT = (bf16t*)wp; wp += (size_t)DD * DD * 2;
  bf16t* WmT = (bf16t*)wp; wp += (size_t)DD * DD * 2;
  bf16t* W0T = (bf16t*)wp; wp += (size_t)D2 * D2 * 2;
  bf16t* W2T = (bf16t*)wp; wp += (size_t)D2 * DD * 2;
  bf16t* kvT = (bf16t*)wp; wp += (size_t)BB * HH * DHH * DHH * 2;
  float* zb  = (float*)wp; wp += (size_t)BB * HH * DHH * 4;
  float* nrm = (float*)wp; wp += (size_t)RR * HH * 4;
  // buffer reuse (lifetimes disjoint):
  bf16t* msg_bf    = x_bf;    // after q GEMM, x_bf is dead
  bf16t* msgWm_bf  = src_bf;  // after k/v GEMMs, src_bf is dead
  bf16t* normed_bf = k_bf;    // after transpose+z, k_bf is dead
  bf16t* h_bf      = kT;      // spans kT+vT = 78.6 MB, dead after kv GEMM and z

  // 1) f32 -> bf16 activations; weights -> col-major bf16
  f2b_kernel<<<RR * DD / 256, 256, 0, stream>>>(x, x_bf, (size_t)RR * DD);
  f2b_kernel<<<RR * DD / 256, 256, 0, stream>>>(src, src_bf, (size_t)RR * DD);
  wconv_kernel<<<(DD * DD + 255) / 256, 256, 0, stream>>>(Wq, WqT, DD, DD);
  wconv_kernel<<<(DD * DD + 255) / 256, 256, 0, stream>>>(Wk, WkT, DD, DD);
  wconv_kernel<<<(DD * DD + 255) / 256, 256, 0, stream>>>(Wv, WvT, DD, DD);
  wconv_kernel<<<(DD * DD + 255) / 256, 256, 0, stream>>>(Wm, WmT, DD, DD);
  wconv_kernel<<<(D2 * D2 + 255) / 256, 256, 0, stream>>>(W0, W0T, D2, D2);
  wconv_kernel<<<(D2 * DD + 255) / 256, 256, 0, stream>>>(W2, W2T, D2, DD);

  // 2) q/k/v projections (elu+1 fused for q,k); block tile 256 rows x 64 cols
  dim3 gqkv(RR / 256, DD / 64);
  gemm_wmma<1, true, false><<<gqkv, 256, 0, stream>>>(x_bf, x_bf, DD, DD, DD, WqT, bq, q_bf, nullptr, DD, DD);
  gemm_wmma<1, true, false><<<gqkv, 256, 0, stream>>>(src_bf, src_bf, DD, DD, DD, WkT, bk, k_bf, nullptr, DD, DD);
  gemm_wmma<0, true, false><<<gqkv, 256, 0, stream>>>(src_bf, src_bf, DD, DD, DD, WvT, bv, v_bf, nullptr, DD, DD);

  // 3) per-head transposes for the K=M reduction GEMM
  transpose_head<<<dim3(NN / 32, HH, BB), 256, 0, stream>>>(k_bf, kT);
  transpose_head<<<dim3(NN / 32, HH, BB), 256, 0, stream>>>(v_bf, vT);

  // 4) kv^T (WMMA, K=4800), z, normalizer
  kv_wmma<<<BB * HH, 128, 0, stream>>>(vT, kT, kvT);
  z_kernel<<<BB * HH * DHH / 8, 256, 0, stream>>>(kT, zb);
  norm_kernel<<<RR * HH / 256, 256, 0, stream>>>(q_bf, zb, nrm);

  // 5) message = (q@kv)/(norm+eps)  then  message @ Wm + bm
  msg_wmma<<<dim3(NN / 64, HH, BB), 128, 0, stream>>>(q_bf, kvT, nrm, msg_bf);
  gemm_wmma<0, true, false><<<gqkv, 256, 0, stream>>>(msg_bf, msg_bf, DD, DD, DD, WmT, bm, msgWm_bf, nullptr, DD, DD);

  // 6) LN1(x) -> bf16
  ln_kernel<<<RR / 8, 256, 0, stream>>>(x, nullptr, g1, b1, normed_bf, nullptr);

  // 7) MLP: h = relu([normed, msgWm] @ W0 + b0)  (concat via K-split)
  dim3 gh(RR / 256, D2 / 64);
  gemm_wmma<2, true, false><<<gh, 256, 0, stream>>>(normed_bf, msgWm_bf, DD, DD, DD, W0T, b0, h_bf, nullptr, D2, D2);
  // 8) mlp_out = h @ W2 + b2  (f32 -> d_out)
  dim3 go(RR / 256, DD / 64);
  gemm_wmma<0, false, true><<<go, 256, 0, stream>>>(h_bf, h_bf, D2, D2, D2, W2T, b2, nullptr, out, D2, DD);

  // 9) out = LN2(x + mlp_out) in-place on d_out
  ln_kernel<<<RR / 8, 256, 0, stream>>>(x, out, g2, b2n, nullptr, out);
}
